// TriLevelMultiHeadedAttention_49435073577351
// MI455X (gfx1250) — matrix operand order repaired and checked by reading, not compiled, not run on hardware
//
#include <hip/hip_runtime.h>
#include <hip/hip_bf16.h>

// Problem constants (fixed by the reference)
#define BB 4
#define SS 1024
#define DD 1024
#define HH 16
#define DKK 64
#define DTT 256
#define MM (BB * SS)   // 4096

typedef __attribute__((ext_vector_type(16))) __bf16 v16bf;
typedef __attribute__((ext_vector_type(8)))  __bf16 v8bf;
typedef __attribute__((ext_vector_type(4)))  __bf16 v4bf;
typedef __attribute__((ext_vector_type(2)))  __bf16 v2bf;
typedef __attribute__((ext_vector_type(8)))  float  v8f;

static __device__ __forceinline__ __bf16 f2bf(float f) {
  unsigned int u = __builtin_bit_cast(unsigned int, f);
  u += 0x7FFFu + ((u >> 16) & 1u);            // round-to-nearest-even
  unsigned short h = (unsigned short)(u >> 16);
  return __builtin_bit_cast(__bf16, h);
}

static __device__ __forceinline__ v16bf cat16(v8bf lo, v8bf hi) {
  return __builtin_shufflevector(lo, hi, 0,1,2,3,4,5,6,7,8,9,10,11,12,13,14,15);
}

// DPP16 row_ror:<N> move (VALU-only cross-lane within 16-lane rows)
template <int CTRL>
static __device__ __forceinline__ float dpp_ror(float x) {
  return __builtin_bit_cast(float,
      __builtin_amdgcn_update_dpp(0, __builtin_bit_cast(int, x),
                                  CTRL, 0xF, 0xF, true));
}
// Full 16-lane-row reductions via rotate tree (every lane gets the result)
static __device__ __forceinline__ float rowmax16(float x) {
  x = fmaxf(x, dpp_ror<0x121>(x));   // row_ror:1
  x = fmaxf(x, dpp_ror<0x122>(x));   // row_ror:2
  x = fmaxf(x, dpp_ror<0x124>(x));   // row_ror:4
  x = fmaxf(x, dpp_ror<0x128>(x));   // row_ror:8
  return x;
}
static __device__ __forceinline__ float rowsum16(float x) {
  x += dpp_ror<0x121>(x);
  x += dpp_ror<0x122>(x);
  x += dpp_ror<0x124>(x);
  x += dpp_ror<0x128>(x);
  return x;
}

// ---------------------------------------------------------------------------
// One-time converters (f32 -> bf16), and weight transpose-convert
// ---------------------------------------------------------------------------
__global__ __launch_bounds__(256)
void convert_bf16_kernel(const float4* __restrict__ x, v4bf* __restrict__ y, int n4) {
  const int i = blockIdx.x * 256 + threadIdx.x;
  if (i < n4) {
    const float4 v = x[i];
    v4bf b;
    b[0] = f2bf(v.x); b[1] = f2bf(v.y); b[2] = f2bf(v.z); b[3] = f2bf(v.w);
    y[i] = b;
  }
}

// W: (K, 1024) f32  ->  Wt: (1024, K) bf16   (LDS-tiled 32x32 transpose)
__global__ __launch_bounds__(256)
void transpose_convert_bf16(const float* __restrict__ W, __bf16* __restrict__ Wt, int K) {
  __shared__ float tile[32][33];
  const int tx = threadIdx.x & 31;
  const int ty = threadIdx.x >> 5;           // 0..7
  const int k0 = blockIdx.x * 32;
  const int n0 = blockIdx.y * 32;
#pragma unroll
  for (int r = 0; r < 32; r += 8)
    tile[ty + r][tx] = W[(size_t)(k0 + ty + r) * DD + n0 + tx];
  __syncthreads();
#pragma unroll
  for (int r = 0; r < 32; r += 8)
    Wt[(size_t)(n0 + ty + r) * K + k0 + tx] = f2bf(tile[tx][ty + r]);
}

// ---------------------------------------------------------------------------
// GEMM: C[4096, 1024] = A[4096,K](bf16) @ Wt[1024,K](bf16, pre-transposed) + bias
// 128 threads = 4 waves; block tile 128x128, K-step 32; wave tile 64x64.
// Register double-buffered global->LDS copies, all-b128 moves, no conversions.
// OUTMODE 0: f32 row-major (M,N).   OUTMODE 1: bf16 (B,H,S,DK) head-split.
// ---------------------------------------------------------------------------
#define LDS_S 40   // 32 + 8 pad bf16 -> 80B row stride (16B aligned, conflict-free)

template <int OUTMODE>
__global__ __launch_bounds__(128)
void gemm_bf16_wmma(const __bf16* __restrict__ A, const __bf16* __restrict__ Wt,
                    const float* __restrict__ bias, void* __restrict__ outp, int K) {
  __shared__ __bf16 sA[128 * LDS_S];
  __shared__ __bf16 sB[128 * LDS_S];

  const int t    = threadIdx.x;
  const int lane = t & 31;
  const int w    = t >> 5;          // 0..3
  const int wm   = w >> 1;          // 0..1
  const int wn   = w & 1;           // 0..1
  const int lg   = lane >> 4;       // 0/1
  const int lr   = lane & 15;
  const int mBase = blockIdx.y * 128;
  const int nBase = blockIdx.x * 128;

  const __bf16* aRow = A  + (size_t)(mBase + t) * K;
  const __bf16* bRow = Wt + (size_t)(nBase + t) * K;

  const v8f vzero = {0.f,0.f,0.f,0.f,0.f,0.f,0.f,0.f};
  v8f acc[4][4];
#pragma unroll
  for (int i = 0; i < 4; i++)
#pragma unroll
    for (int j = 0; j < 4; j++) acc[i][j] = vzero;

  // prefetch K-step 0 into registers
  v8bf aR[4], bR[4];
#pragma unroll
  for (int i = 0; i < 4; i++) {
    aR[i] = *(const v8bf*)(aRow + i * 8);
    bR[i] = *(const v8bf*)(bRow + i * 8);
  }

  for (int kk = 0; kk < K; kk += 32) {
    if (kk) __syncthreads();                  // prior compute done before overwrite
#pragma unroll
    for (int i = 0; i < 4; i++) {
      *(v8bf*)&sA[t * LDS_S + i * 8] = aR[i];
      *(v8bf*)&sB[t * LDS_S + i * 8] = bR[i];
    }
    __syncthreads();
    if (kk + 32 < K) {                        // prefetch next step while computing
#pragma unroll
      for (int i = 0; i < 4; i++) {
        aR[i] = *(const v8bf*)(aRow + kk + 32 + i * 8);
        bR[i] = *(const v8bf*)(bRow + kk + 32 + i * 8);
      }
    }

    // cache 4 B-fragments, stream A-fragments
    v16bf bF[4];
#pragma unroll
    for (int nt = 0; nt < 4; nt++) {
      const int col = wn * 64 + nt * 16 + lr;
      bF[nt] = cat16(*(const v8bf*)&sB[col * LDS_S + lg * 8],
                     *(const v8bf*)&sB[col * LDS_S + 16 + lg * 8]);
    }
    const int koff = lg * 8;
#pragma unroll
    for (int mt = 0; mt < 4; mt++) {
      const int row = wm * 64 + mt * 16 + lr;
      const v16bf aF = cat16(*(const v8bf*)&sA[row * LDS_S + koff],
                             *(const v8bf*)&sA[row * LDS_S + 16 + koff]);
#pragma unroll
      for (int nt = 0; nt < 4; nt++)
        acc[mt][nt] = __builtin_amdgcn_wmma_f32_16x16x32_bf16(
            false, aF, false, bF[nt], (short)0, acc[mt][nt], false, false);
    }
  }

  // ---- epilogue: + bias, write out (branch-free via template) ----
#pragma unroll
  for (int nt = 0; nt < 4; nt++) {
    const int n = nBase + wn * 64 + nt * 16 + lr;
    const float bvv = bias[n];
#pragma unroll
    for (int mt = 0; mt < 4; mt++) {
      const int mrow0 = mBase + wm * 64 + mt * 16 + lg * 8;
#pragma unroll
      for (int v = 0; v < 8; v++) {
        const float val = acc[mt][nt][v] + bvv;
        const int m = mrow0 + v;
        if (OUTMODE == 0) {
          ((float*)outp)[(size_t)m * DD + n] = val;
        } else {
          const int b = m >> 10, s = m & (SS - 1);
          const int h = n >> 6, dk = n & (DKK - 1);
          ((__bf16*)outp)[((((size_t)b * HH + h) * SS) + s) * DKK + dk] = f2bf(val);
        }
      }
    }
  }
}

// ---------------------------------------------------------------------------
// Flash-style attention for one (b,h): Q,K,V bf16 in (B,H,S,DK).
// 64 query rows per block (4 waves x 16 rows), 64-key streaming blocks.
// Adds softmax(QK^T/8 with masked->1e-8) @ V into Oacc (f32, same layout).
// ---------------------------------------------------------------------------
#define AS_K 72    // 64 + 8 pad bf16 -> 144B stride

__global__ __launch_bounds__(128)
void attn_wmma(const __bf16* __restrict__ Q, const __bf16* __restrict__ Kb,
               const __bf16* __restrict__ Vb, const int* __restrict__ mask,
               float* __restrict__ Oacc) {
  __shared__ __bf16 sK [64 * AS_K];
  __shared__ __bf16 sVT[64 * AS_K];
  __shared__ __bf16 sP [4 * 16 * AS_K];

  const int t    = threadIdx.x;
  const int lane = t & 31;
  const int w    = t >> 5;           // 0..3
  const int lg   = lane >> 4;
  const int lr   = lane & 15;
  const int bh   = blockIdx.y;       // 0..63
  const int b    = bh >> 4;
  const size_t base = (size_t)bh * SS * DKK;
  const int qBase = blockIdx.x * 64;
  const int* mrow_ptr = mask + b * SS;

  // Q fragments for this wave's 16 rows (DK=64 -> 2 K-steps of 32)
  v16bf qF[2];
  {
    const int qrow = qBase + w * 16 + lr;
    const __bf16* qp = Q + base + (size_t)qrow * DKK;
    const int koff = lg * 8;
#pragma unroll
    for (int s = 0; s < 2; s++) {
      v8bf lo = *(const v8bf*)(qp + s * 32 + koff);
      v8bf hi = *(const v8bf*)(qp + s * 32 + 16 + koff);
      qF[s] = cat16(lo, hi);
    }
  }

  const v8f vzero = {0.f,0.f,0.f,0.f,0.f,0.f,0.f,0.f};
  v8f accO[4];
#pragma unroll
  for (int i = 0; i < 4; i++) accO[i] = vzero;
  float mrow[8], lrow[8];
#pragma unroll
  for (int v = 0; v < 8; v++) { mrow[v] = -INFINITY; lrow[v] = 0.f; }

  for (int j = 0; j < SS / 64; j++) {
    __syncthreads();
    const int c = (t & 7) * 8;        // 0..56 (dk / k chunk)
    // ---- fill K tile (row-major), b128 copies ----
    {
      const int rb = t >> 3;          // 0..15
#pragma unroll
      for (int p = 0; p < 4; p++) {
        const int row = rb + p * 16;
        *(v8bf*)&sK[row * AS_K + c] =
            *(const v8bf*)(Kb + base + (size_t)(j * 64 + row) * DKK + c);
      }
    }
    // ---- fill V^T tile: 2 adjacent key rows -> packed b32 stores ----
    {
      const int kp = (t >> 3) * 2;    // even key 0..30
#pragma unroll
      for (int p = 0; p < 2; p++) {
        const int key0 = kp + p * 32;
        const size_t g0 = base + (size_t)(j * 64 + key0) * DKK + c;
        v8bf v0 = *(const v8bf*)(Vb + g0);
        v8bf v1 = *(const v8bf*)(Vb + g0 + DKK);
#pragma unroll
        for (int e = 0; e < 8; e++) {
          v2bf pr; pr[0] = v0[e]; pr[1] = v1[e];
          *(v2bf*)&sVT[(c + e) * AS_K + key0] = pr;
        }
      }
    }
    __syncthreads();

    // ---- scores = Q @ K^T (16 x 64 per wave) ----
    v8f sc[4];
#pragma unroll
    for (int nt = 0; nt < 4; nt++) sc[nt] = vzero;
#pragma unroll
    for (int s = 0; s < 2; s++) {
#pragma unroll
      for (int nt = 0; nt < 4; nt++) {
        const int key = nt * 16 + lr;
        v8bf lo = *(const v8bf*)&sK[key * AS_K + s * 32 + lg * 8];
        v8bf hi = *(const v8bf*)&sK[key * AS_K + s * 32 + 16 + lg * 8];
        sc[nt] = __builtin_amdgcn_wmma_f32_16x16x32_bf16(
            false, qF[s], false, cat16(lo, hi), (short)0, sc[nt], false, false);
      }
    }

    // ---- scale + reference-style mask (masked -> 1e-8, pre-softmax) ----
    int mv4[4];
#pragma unroll
    for (int nt = 0; nt < 4; nt++) mv4[nt] = mrow_ptr[j * 64 + nt * 16 + lr];
#pragma unroll
    for (int nt = 0; nt < 4; nt++)
#pragma unroll
      for (int v = 0; v < 8; v++) {
        const float x = sc[nt][v] * 0.125f;
        sc[nt][v] = (mv4[nt] == 0) ? 1e-8f : x;
      }

    // ---- online softmax update (rows live in 16-lane halves; DPP reductions) ----
#pragma unroll
    for (int v = 0; v < 8; v++) {
      float mx = fmaxf(fmaxf(sc[0][v], sc[1][v]), fmaxf(sc[2][v], sc[3][v]));
      mx = rowmax16(mx);
      const float mn = fmaxf(mrow[v], mx);
      const float alpha = __expf(mrow[v] - mn);
      float rs = 0.f;
#pragma unroll
      for (int nt = 0; nt < 4; nt++) {
        const float p = __expf(sc[nt][v] - mn);
        sc[nt][v] = p;
        rs += p;
      }
      rs = rowsum16(rs);
      lrow[v] = lrow[v] * alpha + rs;
      mrow[v] = mn;
#pragma unroll
      for (int nd = 0; nd < 4; nd++) accO[nd][v] *= alpha;
    }

    // ---- P -> per-wave LDS (re-layout to A-fragment form), then P @ V ----
    __bf16* pw = &sP[w * 16 * AS_K];
#pragma unroll
    for (int nt = 0; nt < 4; nt++)
#pragma unroll
      for (int v = 0; v < 8; v++)
        pw[(v + 8 * lg) * AS_K + nt * 16 + lr] = f2bf(sc[nt][v]);
    // same-wave LDS ops are in-order; compiler inserts the dscnt wait.
#pragma unroll
    for (int s = 0; s < 2; s++) {
      const int koff = lg * 8;
      v8bf plo = *(const v8bf*)&pw[lr * AS_K + s * 32 + koff];
      v8bf phi = *(const v8bf*)&pw[lr * AS_K + s * 32 + 16 + koff];
      v16bf pF = cat16(plo, phi);
#pragma unroll
      for (int nd = 0; nd < 4; nd++) {
        const int dk = nd * 16 + lr;
        v8bf blo = *(const v8bf*)&sVT[dk * AS_K + s * 32 + lg * 8];
        v8bf bhi = *(const v8bf*)&sVT[dk * AS_K + s * 32 + 16 + lg * 8];
        accO[nd] = __builtin_amdgcn_wmma_f32_16x16x32_bf16(
            false, pF, false, cat16(blo, bhi), (short)0, accO[nd], false, false);
      }
    }
  }

  // ---- normalize and accumulate into Oacc ----
#pragma unroll
  for (int nd = 0; nd < 4; nd++) {
    const int dk = nd * 16 + lr;
#pragma unroll
    for (int v = 0; v < 8; v++) {
      const int qrow = qBase + w * 16 + v + 8 * lg;
      float* p = Oacc + base + (size_t)qrow * DKK + dk;
      *p += accO[nd][v] / lrow[v];
    }
  }
}

// ---------------------------------------------------------------------------
__global__ void zero_f32_kernel(float* __restrict__ p, int n) {
  const int i = blockIdx.x * blockDim.x + threadIdx.x;
  if (i < n) p[i] = 0.f;
}

// Ocomb[(b*S+s)*D + h*64+dk] = bf16(0.5 * Oacc[((b*H+h)*S+s)*64+dk])
__global__ void combine_heads_kernel(const float* __restrict__ Oacc,
                                     __bf16* __restrict__ Ocomb) {
  const int i = blockIdx.x * blockDim.x + threadIdx.x;  // 0 .. B*S*D-1
  const int dk = i & 63;
  const int h  = (i >> 6) & 15;
  const int s  = (i >> 10) & 1023;
  const int b  = i >> 20;
  Ocomb[i] = f2bf(0.5f * Oacc[((((size_t)b * HH + h) * SS) + s) * DKK + dk]);
}

// ---------------------------------------------------------------------------
extern "C" void kernel_launch(void* const* d_in, const int* in_sizes, int n_in,
                              void* d_out, int out_size, void* d_ws, size_t ws_size,
                              hipStream_t stream) {
  (void)in_sizes; (void)n_in; (void)out_size; (void)ws_size;
  const float* x1  = (const float*)d_in[0];
  const float* x2  = (const float*)d_in[1];
  const float* x3  = (const float*)d_in[2];
  const int*   msk = (const int*)  d_in[3];
  const float* Wq  = (const float*)d_in[4];  const float* bq  = (const float*)d_in[5];
  const float* Wk  = (const float*)d_in[6];  const float* bk  = (const float*)d_in[7];
  const float* Wv  = (const float*)d_in[8];  const float* bv  = (const float*)d_in[9];
  const float* Wo  = (const float*)d_in[10]; const float* bo  = (const float*)d_in[11];
  const float* Wtq = (const float*)d_in[12]; const float* btq = (const float*)d_in[13];
  const float* Wtk = (const float*)d_in[14]; const float* btk = (const float*)d_in[15];
  const float* Wtv = (const float*)d_in[16]; const float* btv = (const float*)d_in[17];

  const size_t NE = (size_t)BB * SS * DD;       // 4M elements
  const size_t WN = (size_t)DD * DD;            // 1M
  const size_t WTN = (size_t)DTT * DD;          // 256K

  __bf16* p   = (__bf16*)d_ws;
  __bf16* qb  = p; p += NE;
  __bf16* kb  = p; p += NE;
  __bf16* vb  = p; p += NE;
  __bf16* x1b = p; p += NE;
  __bf16* x2b = p; p += NE;
  __bf16* x3b = p; p += NE / 4;
  __bf16* WqT = p; p += WN;
  __bf16* WkT = p; p += WN;
  __bf16* WvT = p; p += WN;
  __bf16* WoT = p; p += WN;
  __bf16* WtqT = p; p += WTN;
  __bf16* WtkT = p; p += WTN;
  __bf16* WtvT = p; p += WTN;
  __bf16* Ocomb = p; p += NE;
  float* Oacc = (float*)p;

  const dim3 ggrid(DD / 128, MM / 128);         // (8, 32)
  const dim3 agrid(SS / 64, BB * HH);           // (16, 64)
  const dim3 tgridD(DD / 32, DD / 32);          // (32, 32)
  const dim3 tgridT(DTT / 32, DD / 32);         // (8, 32)

  // one-time bf16 conversions / weight transposes
  convert_bf16_kernel<<<(int)(NE / 4 / 256), 256, 0, stream>>>((const float4*)x1, (v4bf*)x1b, (int)(NE / 4));
  convert_bf16_kernel<<<(int)(NE / 4 / 256), 256, 0, stream>>>((const float4*)x2, (v4bf*)x2b, (int)(NE / 4));
  convert_bf16_kernel<<<(int)(NE / 16 / 256), 256, 0, stream>>>((const float4*)x3, (v4bf*)x3b, (int)(NE / 16));
  transpose_convert_bf16<<<tgridD, 256, 0, stream>>>(Wq, WqT, DD);
  transpose_convert_bf16<<<tgridD, 256, 0, stream>>>(Wk, WkT, DD);
  transpose_convert_bf16<<<tgridD, 256, 0, stream>>>(Wv, WvT, DD);
  transpose_convert_bf16<<<tgridD, 256, 0, stream>>>(Wo, WoT, DD);
  transpose_convert_bf16<<<tgridT, 256, 0, stream>>>(Wtq, WtqT, DTT);
  transpose_convert_bf16<<<tgridT, 256, 0, stream>>>(Wtk, WtkT, DTT);
  transpose_convert_bf16<<<tgridT, 256, 0, stream>>>(Wtv, WtvT, DTT);
  zero_f32_kernel<<<(int)(NE / 256), 256, 0, stream>>>(Oacc, (int)NE);

  // stream r : x1
  gemm_bf16_wmma<1><<<ggrid, 128, 0, stream>>>(x1b, WqT, bq, (void*)qb, DD);
  gemm_bf16_wmma<1><<<ggrid, 128, 0, stream>>>(x1b, WkT, bk, (void*)kb, DD);
  gemm_bf16_wmma<1><<<ggrid, 128, 0, stream>>>(x1b, WvT, bv, (void*)vb, DD);
  attn_wmma<<<agrid, 128, 0, stream>>>(qb, kb, vb, msk, Oacc);

  // stream n : x2
  gemm_bf16_wmma<1><<<ggrid, 128, 0, stream>>>(x2b, WqT, bq, (void*)qb, DD);
  gemm_bf16_wmma<1><<<ggrid, 128, 0, stream>>>(x2b, WkT, bk, (void*)kb, DD);
  gemm_bf16_wmma<1><<<ggrid, 128, 0, stream>>>(x2b, WvT, bv, (void*)vb, DD);
  attn_wmma<<<agrid, 128, 0, stream>>>(qb, kb, vb, msk, Oacc);

  // stream t : x3 (K = DT = 256)
  gemm_bf16_wmma<1><<<ggrid, 128, 0, stream>>>(x3b, WtqT, btq, (void*)qb, DTT);
  gemm_bf16_wmma<1><<<ggrid, 128, 0, stream>>>(x3b, WtkT, btk, (void*)kb, DTT);
  gemm_bf16_wmma<1><<<ggrid, 128, 0, stream>>>(x3b, WtvT, btv, (void*)vb, DTT);
  attn_wmma<<<agrid, 128, 0, stream>>>(qb, kb, vb, msk, Oacc);

  // combine (x = 0.5*(xr+xn+xt), heads -> (B,S,D)) then output projection
  combine_heads_kernel<<<(int)(NE / 256), 256, 0, stream>>>(Oacc, Ocomb);
  gemm_bf16_wmma<0><<<ggrid, 128, 0, stream>>>(Ocomb, WoT, bo, d_out, DD);
}
